// MultiHeadSelfAttention_25134148616885
// MI455X (gfx1250) — matrix operand order repaired and checked by reading, not compiled, not run on hardware
//
#include <hip/hip_runtime.h>

// ---------------------------------------------------------------------------
// Types for CDNA5 WMMA (wave32, 16x16x32 bf16 -> f32)
// ---------------------------------------------------------------------------
typedef __attribute__((ext_vector_type(16))) __bf16 v16bf;
typedef __attribute__((ext_vector_type(8)))  float  v8f;

union FragU { unsigned int u[8]; v16bf v; };

__device__ __forceinline__ v8f wmma_bf16(v16bf a, v16bf b, v8f c) {
    // (neg_a, A, neg_b, B, c_mod, C, reuse_a, reuse_b)
    return __builtin_amdgcn_wmma_f32_16x16x32_bf16(false, a, false, b, (short)0, c,
                                                   false, false);
}

#define Bsz 4
#define Ssz 2048
#define Hsz 1024
#define NHh 16
#define HDd 64
#define NEGF (-3.4028235e38f)

// padded LDS row widths (keep 16B alignment, spread rows across banks)
#define PAD40 40   // 80B rows  -> 20-bank stride
#define PAD72 72   // 144B rows -> 36-bank stride

// ---------------------------------------------------------------------------
// fp32 -> bf16 convert
// ---------------------------------------------------------------------------
__global__ void cvt_bf16_kernel(const float* __restrict__ in,
                                __bf16* __restrict__ out, int n) {
    int i = blockIdx.x * blockDim.x + threadIdx.x;
    int stride = gridDim.x * blockDim.x;
    for (; i < n; i += stride) out[i] = (__bf16)in[i];
}

// ---------------------------------------------------------------------------
// bf16 GEMM  Y = A(MxK) @ W(KxN) + bias, M=8192, N=K=1024
// MODE 0: write bf16 in (B, NH, S, HD) head layout   (QKV projections)
// MODE 1: write f32 = Y + resid                      (output projection)
// block = 128 threads (4 waves); block tile 128x64, wave tile 32x64
// Weight tile staged TRANSPOSED in LDS so B-fragments are 2x ds_load_b128.
// ---------------------------------------------------------------------------
template <int MODE>
__global__ __launch_bounds__(128)
void gemm_bf16_kernel(const __bf16* __restrict__ A, const __bf16* __restrict__ W,
                      const float* __restrict__ bias, const float* __restrict__ resid,
                      __bf16* __restrict__ out_bf, float* __restrict__ out_f32) {
    __shared__ __align__(16) __bf16 As[128][PAD40];  // [m][k], k 0..31 valid
    __shared__ __align__(16) __bf16 Bt[64][PAD40];   // [n][k] transposed

    const int tid   = threadIdx.x;
    const int lane  = tid & 31;
    const int wv    = tid >> 5;
    const int mbase = blockIdx.x * 128;
    const int nbase = blockIdx.y * 64;

    v8f acc[2][4];
#pragma unroll
    for (int u = 0; u < 2; ++u)
#pragma unroll
        for (int c = 0; c < 4; ++c)
#pragma unroll
            for (int i = 0; i < 8; ++i) acc[u][c][i] = 0.f;

    const int brow = tid >> 2, bq = (tid & 3) * 16;
    const int am   = lane & 15;
    const int kh   = (lane >> 4) << 2;   // A-frag uint offset: 0 or 4
    const int kb16 = (lane >> 4) << 4;   // unused; B-frags index with kh
    const int ncl  = lane & 15;

    for (int kk = 0; kk < Hsz; kk += 32) {
        // stage A: one 32-elem row (64B) per thread
        {
            const uint4* as = (const uint4*)(A + (size_t)(mbase + tid) * Hsz + kk);
            uint4* ad = (uint4*)&As[tid][0];
            ad[0] = as[0]; ad[1] = as[1]; ad[2] = as[2]; ad[3] = as[3];
        }
        // stage W transposed: 16 consecutive n for one k row per thread
        {
            union { uint4 q[2]; __bf16 h[16]; } tw;
            const uint4* bs = (const uint4*)(W + (size_t)(kk + brow) * Hsz + nbase + bq);
            tw.q[0] = bs[0]; tw.q[1] = bs[1];
#pragma unroll
            for (int ii = 0; ii < 16; ++ii) Bt[bq + ii][brow] = tw.h[ii];
        }
        __syncthreads();

        // two A fragments (16x32) per wave, dword-packed (2x b128 each)
        FragU fa[2];
#pragma unroll
        for (int u = 0; u < 2; ++u) {
            const unsigned int* arp = (const unsigned int*)&As[wv * 32 + u * 16 + am][0];
#pragma unroll
            for (int i = 0; i < 4; ++i) {
                fa[u].u[i] = arp[kh + i];
                fa[u].u[4 + i] = arp[8 + kh + i];
            }
        }

#pragma unroll
        for (int c = 0; c < 4; ++c) {
            FragU fb;
            const unsigned int* brp = (const unsigned int*)&Bt[c * 16 + ncl][0];
#pragma unroll
            for (int i = 0; i < 8; ++i) fb.u[i] = brp[(i >> 2) * 8 + kh + (i & 3)];
#pragma unroll
            for (int u = 0; u < 2; ++u)
                acc[u][c] = wmma_bf16(fa[u].v, fb.v, acc[u][c]);
        }
        __syncthreads();
    }

    // epilogue (straight-line per instantiation)
    const int r0 = (lane >> 4) * 8;
    if (MODE == 0) {
#pragma unroll
        for (int c = 0; c < 4; ++c) {
            const int ncol = nbase + c * 16 + ncl;
            const float bvv = bias[ncol];
            const int hh = ncol >> 6, d = ncol & 63;
#pragma unroll
            for (int u = 0; u < 2; ++u) {
#pragma unroll
                for (int v = 0; v < 8; ++v) {
                    const int row = mbase + wv * 32 + u * 16 + r0 + v;
                    const int b = row >> 11, s = row & 2047;
                    out_bf[(((size_t)(b * NHh + hh)) * Ssz + s) * HDd + d] =
                        (__bf16)(acc[u][c][v] + bvv);
                }
            }
        }
    } else {
        // batch residual loads, then add+store
        float rsd[4][2][8];
#pragma unroll
        for (int c = 0; c < 4; ++c) {
            const int ncol = nbase + c * 16 + ncl;
#pragma unroll
            for (int u = 0; u < 2; ++u)
#pragma unroll
                for (int v = 0; v < 8; ++v) {
                    const int row = mbase + wv * 32 + u * 16 + r0 + v;
                    rsd[c][u][v] = resid[(size_t)row * Hsz + ncol];
                }
        }
#pragma unroll
        for (int c = 0; c < 4; ++c) {
            const int ncol = nbase + c * 16 + ncl;
            const float bvv = bias[ncol];
#pragma unroll
            for (int u = 0; u < 2; ++u)
#pragma unroll
                for (int v = 0; v < 8; ++v) {
                    const int row = mbase + wv * 32 + u * 16 + r0 + v;
                    out_f32[(size_t)row * Hsz + ncol] = acc[u][c][v] + bvv + rsd[c][u][v];
                }
        }
    }
}

// ---------------------------------------------------------------------------
// Flash-style attention: grid (S/64, B*NH), block 128 (4 waves)
// wave = 16 query rows; loop over 32-key tiles; online softmax in registers
// ---------------------------------------------------------------------------
__global__ __launch_bounds__(128)
void attn_kernel(const __bf16* __restrict__ qb, const __bf16* __restrict__ kbuf,
                 const __bf16* __restrict__ vbuf, const float* __restrict__ mask,
                 __bf16* __restrict__ ctx) {
    __shared__ __align__(16) __bf16 Ks[32][PAD72];     // [key][d], d 0..63 valid
    __shared__ __align__(16) __bf16 Vt[64][PAD40];     // [d][key] (transposed)
    __shared__ __align__(16) __bf16 Ps[4][16][PAD40];  // per-wave P tile

    const int tid   = threadIdx.x;
    const int lane  = tid & 31;
    const int wv    = tid >> 5;
    const int bh    = blockIdx.y;            // b*NH + h
    const int b     = bh >> 4;
    const int h     = bh & 15;
    const int qrow0 = blockIdx.x * 64 + wv * 16;

    const int am   = lane & 15;
    const int kh   = (lane >> 4) << 2;
    const int ncl  = lane & 15;
    const int kb16 = (lane >> 4) << 4;

    // resident Q fragments: 16x64 = two 16x32 A-frags
    FragU fq0, fq1;
    {
        const unsigned int* qp =
            (const unsigned int*)(qb + ((size_t)bh * Ssz + qrow0 + am) * HDd);
#pragma unroll
        for (int i = 0; i < 4; ++i) {
            fq0.u[i]     = qp[kh + i];      fq0.u[4 + i] = qp[8 + kh + i];
            fq1.u[i]     = qp[16 + kh + i]; fq1.u[4 + i] = qp[24 + kh + i];
        }
    }

    v8f o[4];
#pragma unroll
    for (int c = 0; c < 4; ++c)
#pragma unroll
        for (int i = 0; i < 8; ++i) o[c][i] = 0.f;
    float mrow[8], lrow[8];
#pragma unroll
    for (int v = 0; v < 8; ++v) { mrow[v] = NEGF; lrow[v] = 0.f; }

    const int ldrow = tid >> 2, ldcol = (tid & 3) * 16;

    for (int j = 0; j < Ssz; j += 32) {
        // stage K (row-major, padded) and V (transposed, padded) tiles
        {
            const uint4* ksrc =
                (const uint4*)(kbuf + ((size_t)bh * Ssz + j + ldrow) * HDd + ldcol);
            uint4* kd = (uint4*)&Ks[ldrow][ldcol];
            kd[0] = ksrc[0]; kd[1] = ksrc[1];

            union { uint4 q[2]; __bf16 hh[16]; } tv;
            const uint4* vsrc =
                (const uint4*)(vbuf + ((size_t)bh * Ssz + j + ldrow) * HDd + ldcol);
            tv.q[0] = vsrc[0]; tv.q[1] = vsrc[1];
#pragma unroll
            for (int ii = 0; ii < 16; ++ii) Vt[ldcol + ii][ldrow] = tv.hh[ii];
        }
        __syncthreads();

        // scores: two 16x16 key tiles, contraction 64 split over two WMMAs
        v8f s[2];
#pragma unroll
        for (int t = 0; t < 2; ++t) {
            const unsigned int* krp = (const unsigned int*)&Ks[t * 16 + ncl][0];
            FragU fk0, fk1;
#pragma unroll
            for (int i = 0; i < 8; ++i) {
                fk0.u[i] = krp[(i >> 2) * 8 + kh + (i & 3)];
                fk1.u[i] = krp[16 + (i >> 2) * 8 + kh + (i & 3)];
            }
            v8f z;
#pragma unroll
            for (int i = 0; i < 8; ++i) z[i] = 0.f;
            z = wmma_bf16(fq0.v, fk0.v, z);
            z = wmma_bf16(fq1.v, fk1.v, z);
            const float msk  = mask[(size_t)b * Ssz + j + t * 16 + ncl];
            const float madd = (1.0f - msk) * NEGF;
#pragma unroll
            for (int i = 0; i < 8; ++i) s[t][i] = z[i] * 0.125f + madd;
        }

        // online softmax per row (rows live in VGPR index; cols across 16 lanes)
#pragma unroll
        for (int v = 0; v < 8; ++v) {
            float lm = fmaxf(s[0][v], s[1][v]);
#pragma unroll
            for (int d = 1; d < 16; d <<= 1) lm = fmaxf(lm, __shfl_xor(lm, d, 32));
            const float mnew  = fmaxf(mrow[v], lm);
            const float alpha = __expf(mrow[v] - mnew);
            const float p0 = __expf(s[0][v] - mnew);
            const float p1 = __expf(s[1][v] - mnew);
            float rs = p0 + p1;
#pragma unroll
            for (int d = 1; d < 16; d <<= 1) rs += __shfl_xor(rs, d, 32);
            lrow[v] = lrow[v] * alpha + rs;
            mrow[v] = mnew;
#pragma unroll
            for (int c = 0; c < 4; ++c) o[c][v] *= alpha;
            s[0][v] = p0; s[1][v] = p1;
        }

        // transpose P (C-layout -> row-major bf16) through per-wave LDS
        {
            const int pr0 = (lane >> 4) * 8;
#pragma unroll
            for (int v = 0; v < 8; ++v) {
                Ps[wv][pr0 + v][ncl]      = (__bf16)s[0][v];
                Ps[wv][pr0 + v][16 + ncl] = (__bf16)s[1][v];
            }
        }
        __syncthreads();

        // PV: A = P(16x32), B = V chunk (32x16) from transposed tile
        {
            FragU fp;
            const unsigned int* prp = (const unsigned int*)&Ps[wv][am][0];
#pragma unroll
            for (int i = 0; i < 4; ++i) { fp.u[i] = prp[kh + i]; fp.u[4 + i] = prp[8 + kh + i]; }
#pragma unroll
            for (int c = 0; c < 4; ++c) {
                const unsigned int* vrp = (const unsigned int*)&Vt[c * 16 + ncl][0];
                FragU fv;
#pragma unroll
                for (int i = 0; i < 8; ++i) fv.u[i] = vrp[(i >> 2) * 8 + kh + (i & 3)];
                o[c] = wmma_bf16(fp.v, fv.v, o[c]);
            }
        }
        __syncthreads();
    }

    // normalize and store ctx in flat (B,S,H) bf16 layout
    const int r0 = (lane >> 4) * 8;
#pragma unroll
    for (int v = 0; v < 8; ++v) {
        const float inv = 1.0f / lrow[v];
        const int srow = blockIdx.x * 64 + wv * 16 + r0 + v;
#pragma unroll
        for (int c = 0; c < 4; ++c) {
            const int col = h * HDd + c * 16 + ncl;
            ctx[((size_t)b * Ssz + srow) * Hsz + col] = (__bf16)(o[c][v] * inv);
        }
    }
}

// ---------------------------------------------------------------------------
// LayerNorm: one block per row of 1024
// ---------------------------------------------------------------------------
__global__ __launch_bounds__(256)
void ln_kernel(const float* __restrict__ x, const float* __restrict__ g,
               const float* __restrict__ bta, float* __restrict__ out) {
    const int row = blockIdx.x;
    const float* xr = x + (size_t)row * Hsz;
    float s = 0.f, s2 = 0.f;
    for (int i = threadIdx.x; i < Hsz; i += 256) {
        const float v = xr[i];
        s += v; s2 += v * v;
    }
    __shared__ float red0[256], red1[256];
    red0[threadIdx.x] = s; red1[threadIdx.x] = s2;
    __syncthreads();
    for (int off = 128; off > 0; off >>= 1) {
        if (threadIdx.x < off) {
            red0[threadIdx.x] += red0[threadIdx.x + off];
            red1[threadIdx.x] += red1[threadIdx.x + off];
        }
        __syncthreads();
    }
    const float mu  = red0[0] * (1.0f / Hsz);
    const float var = red1[0] * (1.0f / Hsz) - mu * mu;
    const float inv = rsqrtf(var + 1e-12f);
    for (int i = threadIdx.x; i < Hsz; i += 256)
        out[(size_t)row * Hsz + i] = (xr[i] - mu) * inv * g[i] + bta[i];
}

// ---------------------------------------------------------------------------
// launcher
// ---------------------------------------------------------------------------
extern "C" void kernel_launch(void* const* d_in, const int* in_sizes, int n_in,
                              void* d_out, int out_size, void* d_ws, size_t ws_size,
                              hipStream_t stream) {
    (void)in_sizes; (void)n_in; (void)out_size; (void)ws_size;
    const float* hs   = (const float*)d_in[0];
    const float* mask = (const float*)d_in[1];
    const float* Wq   = (const float*)d_in[2];
    const float* bq   = (const float*)d_in[3];
    const float* Wk   = (const float*)d_in[4];
    const float* bk   = (const float*)d_in[5];
    const float* Wv   = (const float*)d_in[6];
    const float* bv   = (const float*)d_in[7];
    const float* Wo   = (const float*)d_in[8];
    const float* bo   = (const float*)d_in[9];
    const float* lng  = (const float*)d_in[10];
    const float* lnb  = (const float*)d_in[11];

    const size_t M  = (size_t)Bsz * Ssz;      // 8192
    const size_t XE = M * Hsz;                // 8.39M elems
    const size_t WE = (size_t)Hsz * Hsz;      // 1.05M elems

    char* ws = (char*)d_ws;
    size_t off = 0;
    __bf16* Xbf = (__bf16*)(ws + off); off += XE * 2;
    __bf16* Wqb = (__bf16*)(ws + off); off += WE * 2;
    __bf16* Wkb = (__bf16*)(ws + off); off += WE * 2;
    __bf16* Wvb = (__bf16*)(ws + off); off += WE * 2;
    __bf16* Wob = (__bf16*)(ws + off); off += WE * 2;
    __bf16* Qbf = (__bf16*)(ws + off); off += XE * 2;
    __bf16* Kbf = (__bf16*)(ws + off); off += XE * 2;
    __bf16* Vbf = (__bf16*)(ws + off); off += XE * 2;
    __bf16* Cbf = (__bf16*)(ws + off); off += XE * 2;
    float*  Xres = (float*)(ws + off);  off += XE * 4;

    // fp32 -> bf16 converts
    cvt_bf16_kernel<<<2048, 256, 0, stream>>>(hs, Xbf, (int)XE);
    cvt_bf16_kernel<<<512, 256, 0, stream>>>(Wq, Wqb, (int)WE);
    cvt_bf16_kernel<<<512, 256, 0, stream>>>(Wk, Wkb, (int)WE);
    cvt_bf16_kernel<<<512, 256, 0, stream>>>(Wv, Wvb, (int)WE);
    cvt_bf16_kernel<<<512, 256, 0, stream>>>(Wo, Wob, (int)WE);

    // QKV projections -> head-layout bf16
    dim3 gg(64, 16);
    gemm_bf16_kernel<0><<<gg, 128, 0, stream>>>(Xbf, Wqb, bq, nullptr, Qbf, nullptr);
    gemm_bf16_kernel<0><<<gg, 128, 0, stream>>>(Xbf, Wkb, bk, nullptr, Kbf, nullptr);
    gemm_bf16_kernel<0><<<gg, 128, 0, stream>>>(Xbf, Wvb, bv, nullptr, Vbf, nullptr);

    // flash attention
    attn_kernel<<<dim3(Ssz / 64, Bsz * NHh), 128, 0, stream>>>(Qbf, Kbf, Vbf, mask, Cbf);

    // output projection + bias + residual (f32)
    gemm_bf16_kernel<1><<<gg, 128, 0, stream>>>(Cbf, Wob, bo, hs, nullptr, Xres);

    // layernorm -> d_out
    ln_kernel<<<(int)M, 256, 0, stream>>>(Xres, lng, lnb, (float*)d_out);
}
